// MultiScaleAttentionBlock_19885698580541
// MI455X (gfx1250) — matrix operand order repaired and checked by reading, not compiled, not run on hardware
//
#include <hip/hip_runtime.h>
#include <hip/hip_bf16.h>
#include <math.h>

// ---------------- types for WMMA (gfx1250, wave32) ----------------
typedef __attribute__((ext_vector_type(16))) __bf16 v16bf;
typedef __attribute__((ext_vector_type(8)))  __bf16 v8bf;
typedef __attribute__((ext_vector_type(8)))  float  v8f;
// exact pointee type expected by the gfx1250 async-to-LDS builtin (from hipcc diag)
typedef int v4i_vec __attribute__((vector_size(16)));

#define E_      256
#define HEADS_  8
#define HD_     32
#define B_      8
#define H_      128
#define W_      128
#define HW_     16384
#define PIX_    131072   // B*H*W
#define NWIN_   2048
#define LW_     64
#define CCAT_   512
#define BN_EPS_ 1e-5f

// ---------------- fragment helpers ----------------
__device__ __forceinline__ v16bf join16(v8bf lo, v8bf hi) {
  v16bf r;
#pragma unroll
  for (int i = 0; i < 8; ++i) { r[i] = lo[i]; r[8 + i] = hi[i]; }
  return r;
}

// B operand / any 16-contiguous-bf16 fragment (lane holds 16 consecutive K)
__device__ __forceinline__ v16bf load_frag16(const __bf16* base) {
  const v8bf* p = (const v8bf*)base;
  return join16(p[0], p[1]);
}

// A operand, 16x32 bf16, row-major row with contiguous K.
// Contiguous layout: lane group g: halves 0..15 -> K = g*16 + j
__device__ __forceinline__ v16bf load_arow(const __bf16* rowbase, int g) {
  const v8bf* lo = (const v8bf*)(rowbase + g * 16);
  const v8bf* hi = (const v8bf*)(rowbase + g * 16 + 8);
  return join16(*lo, *hi);
}

__device__ __forceinline__ v8f wmma_bf16(v16bf a, v16bf b, v8f c) {
  return __builtin_amdgcn_wmma_f32_16x16x32_bf16(false, a, false, b, (short)0, c,
                                                 false, false);
}

__device__ __forceinline__ void zero8(v8f& v) {
#pragma unroll
  for (int i = 0; i < 8; ++i) v[i] = 0.0f;
}

// ---------------- prep kernels ----------------
__global__ void cvt_bf16_kernel(const float* __restrict__ src,
                                __bf16* __restrict__ dst, int n) {
  int i = blockIdx.x * 256 + threadIdx.x;
  if (i < n) dst[i] = (__bf16)src[i];
}

// post_w (256,512,3,3) -> Wc[r][oc][ic] bf16 (r = kh*3+kw)
__global__ void pack_convw_kernel(const float* __restrict__ pw,
                                  __bf16* __restrict__ dst) {
  int i = blockIdx.x * 256 + threadIdx.x;
  if (i >= 9 * 256 * 512) return;
  int ic = i & 511;
  int t = i >> 9;
  int oc = t & 255;
  int r = t >> 8;
  dst[i] = (__bf16)pw[(size_t)(oc * 512 + ic) * 9 + r];
}

// pe[e][h*128+w] f32, 2D sin/cos positional encoding (E=256, d_half=128)
__global__ void pe_kernel(float* __restrict__ pe) {
  int i = blockIdx.x * 256 + threadIdx.x;
  if (i >= E_ * HW_) return;
  int hw = i & (HW_ - 1);
  int e = i >> 14;
  int h = hw >> 7, w = hw & 127;
  int c = e & 127;
  float pos = (e < 128) ? (float)h : (float)w;
  int di = c >> 1;
  float dv = __expf((float)(2 * di) * (-9.210340371976184f / 128.0f));
  float ang = pos * dv;
  pe[i] = (c & 1) ? __cosf(ang) : __sinf(ang);
}

// ---------------- GEMM over channel-major f32 activations ----------------
// out[m][n] = sum_k A[b, k, p] * Bw[n][k]  (+pe[n][p])
// mode 0: bf16 windowed [win][l][256]; mode 1: bf16 concat (b, chan_base+n, p)
__global__ __launch_bounds__(128)
void gemm_colA_kernel(const float* __restrict__ A, int Cin,
                      const __bf16* __restrict__ Bw,
                      const float* __restrict__ pe,
                      __bf16* __restrict__ out, int mode, int chan_base) {
  __shared__ __bf16 As[64 * 32];
  const int tid = threadIdx.x;
  const int lane = tid & 31;
  const int wid = tid >> 5;   // wave 0..3
  const int g = lane >> 4;
  const int ln = lane & 15;

  const int m0 = blockIdx.x * 64;
  const int n0 = blockIdx.y * 64;
  const int b = m0 >> 14;
  const int p0 = m0 & (HW_ - 1);

  v8f acc[4];
#pragma unroll
  for (int j = 0; j < 4; ++j) zero8(acc[j]);

  const int kseg = tid & 31;
  const int mseg = tid >> 5;
  const int nchunks = Cin >> 5;

  for (int kc = 0; kc < nchunks; ++kc) {
    const float* src =
        A + ((size_t)(b * Cin + kc * 32 + kseg)) * HW_ + p0 + mseg * 16;
    __builtin_prefetch(src + 32 * HW_, 0, 0);  // global_prefetch next K chunk
#pragma unroll
    for (int i = 0; i < 16; ++i)
      As[(mseg * 16 + i) * 32 + kseg] = (__bf16)src[i];
    __syncthreads();

    const int mloc = wid * 16 + ln;
    v16bf a = load_arow(&As[mloc * 32], g);
#pragma unroll
    for (int nj = 0; nj < 4; ++nj) {
      const int n = n0 + nj * 16 + ln;
      v16bf bb = load_frag16(Bw + (size_t)n * Cin + kc * 32 + g * 16);
      acc[nj] = wmma_bf16(a, bb, acc[nj]);
    }
    __syncthreads();
  }

#pragma unroll
  for (int nj = 0; nj < 4; ++nj) {
#pragma unroll
    for (int i = 0; i < 8; ++i) {
      int mloc = wid * 16 + g * 8 + i;
      int p = p0 + mloc;
      int n = n0 + nj * 16 + ln;
      float val = acc[nj][i];
      if (pe) val += pe[(size_t)n * HW_ + p];
      if (mode == 0) {
        int h = p >> 7, w = p & 127;
        int win = b * 256 + (h >> 3) * 16 + (w >> 3);
        int l = (h & 7) * 8 + (w & 7);
        out[((size_t)win * LW_ + l) * E_ + n] = (__bf16)val;
      } else {
        out[((size_t)(b * CCAT_ + chan_base + n)) * HW_ + p] = (__bf16)val;
      }
    }
  }
}

// ---------------- GEMM over row-major bf16 A ----------------
// out[m][n] = sum_k A[m][k]*Bw[n][k] + bias[n]
// mode 0: bf16 out[m*ldo + col_off + n]; mode 1: window-reverse into concat buf
__global__ __launch_bounds__(128)
void gemm_rowA_kernel(const __bf16* __restrict__ A, int K,
                      const __bf16* __restrict__ Bw,
                      const float* __restrict__ bias,
                      __bf16* __restrict__ out, int ldo, int col_off, int mode) {
  const int tid = threadIdx.x;
  const int lane = tid & 31;
  const int wid = tid >> 5;
  const int g = lane >> 4;
  const int ln = lane & 15;
  const int m0 = blockIdx.x * 64;
  const int n0 = blockIdx.y * 64;

  v8f acc[4];
#pragma unroll
  for (int j = 0; j < 4; ++j) zero8(acc[j]);

  const int m = m0 + wid * 16 + ln;
  const __bf16* arow = A + (size_t)m * K;
  const int nchunks = K >> 5;

  for (int kc = 0; kc < nchunks; ++kc) {
    __builtin_prefetch(arow + kc * 32 + 32, 0, 0);
    v16bf a = load_arow(arow + kc * 32, g);
#pragma unroll
    for (int nj = 0; nj < 4; ++nj) {
      const int n = n0 + nj * 16 + ln;
      v16bf bb = load_frag16(Bw + (size_t)n * K + kc * 32 + g * 16);
      acc[nj] = wmma_bf16(a, bb, acc[nj]);
    }
  }

#pragma unroll
  for (int nj = 0; nj < 4; ++nj) {
#pragma unroll
    for (int i = 0; i < 8; ++i) {
      int mm = m0 + wid * 16 + g * 8 + i;
      int n = n0 + nj * 16 + ln;
      float val = acc[nj][i] + (bias ? bias[n] : 0.0f);
      if (mode == 0) {
        out[(size_t)mm * ldo + col_off + n] = (__bf16)val;
      } else {
        int win = mm >> 6, l = mm & 63;
        int bb_ = win >> 8, wi = win & 255;
        int h = (wi >> 4) * 8 + (l >> 3);
        int w = (wi & 15) * 8 + (l & 7);
        out[((size_t)(bb_ * CCAT_ + n)) * HW_ + h * 128 + w] = (__bf16)val;
      }
    }
  }
}

// ---------------- windowed multi-head attention ----------------
// qkv: [win][64][768] bf16 (q|k|v), obuf: [win][64][256] bf16
__global__ __launch_bounds__(256)
void attn_kernel(const __bf16* __restrict__ qkv, __bf16* __restrict__ obuf) {
  __shared__ __bf16 Vs[LW_ * E_];          // 32 KB
  __shared__ __bf16 Ps[HEADS_][16 * LW_];  // 16 KB
  const int tid = threadIdx.x;
  const int lane = tid & 31;
  const int head = tid >> 5;  // one wave per head
  const int g = lane >> 4;
  const int ln = lane & 15;
  const int win = blockIdx.x;
  const __bf16* base = qkv + (size_t)win * LW_ * 768;

  {  // stage V (all heads) into LDS: 128 contiguous bytes per thread
    int l = tid >> 2, seg = tid & 3;
    const __bf16* gsrc = base + (size_t)l * 768 + 512 + seg * 64;
    __bf16* ldst = &Vs[l * E_ + seg * 64];
#if defined(__gfx1250__) && __has_builtin(__builtin_amdgcn_global_load_async_to_lds_b128)
    // CDNA5 async global->LDS copy path (ASYNCcnt-tracked)
    __bf16* gnc = const_cast<__bf16*>(gsrc);
#pragma unroll
    for (int i = 0; i < 8; ++i)
      __builtin_amdgcn_global_load_async_to_lds_b128(
          (__attribute__((address_space(1))) v4i_vec*)(gnc + i * 8),
          (__attribute__((address_space(3))) v4i_vec*)(ldst + i * 8), 0, 0);
#if __has_builtin(__builtin_amdgcn_s_wait_asynccnt)
    __builtin_amdgcn_s_wait_asynccnt(0);
#endif
#else
    const v8bf* src = (const v8bf*)gsrc;
    v8bf* dst = (v8bf*)ldst;
#pragma unroll
    for (int i = 0; i < 8; ++i) dst[i] = src[i];
#endif
  }
  __syncthreads();

  // K fragments: B operand for S = Q*K^T; element (K=d, N=token)
  v16bf kf[4];
#pragma unroll
  for (int nj = 0; nj < 4; ++nj) {
    int tok = nj * 16 + ln;
    kf[nj] = load_frag16(base + (size_t)tok * 768 + 256 + head * 32 + g * 16);
  }
  // V fragments: B operand for O = P*V; element (K=token, N=d)
  v16bf vf[2][2];
#pragma unroll
  for (int kc = 0; kc < 2; ++kc)
#pragma unroll
    for (int nd = 0; nd < 2; ++nd) {
      v16bf t;
#pragma unroll
      for (int j = 0; j < 16; ++j) {
        int tok = kc * 32 + g * 16 + j;
        t[j] = Vs[tok * E_ + head * 32 + nd * 16 + ln];
      }
      vf[kc][nd] = t;
    }

  const float scale = 0.17677669529663687f;  // 1/sqrt(32)

  for (int mi = 0; mi < 4; ++mi) {
    int tok = mi * 16 + ln;
    v16bf qa = load_arow(base + (size_t)tok * 768 + head * 32, g);

    v8f s[4];
#pragma unroll
    for (int j = 0; j < 4; ++j) zero8(s[j]);
#pragma unroll
    for (int nj = 0; nj < 4; ++nj) s[nj] = wmma_bf16(qa, kf[nj], s[nj]);

    // row softmax over 64 cols (rows striped across 16-lane halves)
    float pv[4][8];
#pragma unroll
    for (int i = 0; i < 8; ++i) {
      float mx = -1e30f;
#pragma unroll
      for (int nj = 0; nj < 4; ++nj) mx = fmaxf(mx, s[nj][i] * scale);
      for (int off = 8; off; off >>= 1) mx = fmaxf(mx, __shfl_xor(mx, off, 32));
      float sum = 0.0f;
#pragma unroll
      for (int nj = 0; nj < 4; ++nj) {
        float e = __expf(s[nj][i] * scale - mx);
        pv[nj][i] = e;
        sum += e;
      }
      for (int off = 8; off; off >>= 1) sum += __shfl_xor(sum, off, 32);
      float inv = 1.0f / sum;
#pragma unroll
      for (int nj = 0; nj < 4; ++nj) pv[nj][i] *= inv;
    }

    // C-layout -> A-layout transpose of P through per-head LDS scratch
#pragma unroll
    for (int nj = 0; nj < 4; ++nj)
#pragma unroll
      for (int i = 0; i < 8; ++i)
        Ps[head][(g * 8 + i) * LW_ + nj * 16 + ln] = (__bf16)pv[nj][i];
    __syncthreads();

    v8f oacc[2];
#pragma unroll
    for (int j = 0; j < 2; ++j) zero8(oacc[j]);
#pragma unroll
    for (int kc = 0; kc < 2; ++kc) {
      v16bf pa = load_arow(&Ps[head][ln * LW_ + kc * 32], g);
#pragma unroll
      for (int nd = 0; nd < 2; ++nd)
        oacc[nd] = wmma_bf16(pa, vf[kc][nd], oacc[nd]);
    }
#pragma unroll
    for (int nd = 0; nd < 2; ++nd)
#pragma unroll
      for (int i = 0; i < 8; ++i) {
        int t2 = mi * 16 + g * 8 + i;
        int d = nd * 16 + ln;
        obuf[((size_t)win * LW_ + t2) * E_ + head * 32 + d] = (__bf16)oacc[nd][i];
      }
    __syncthreads();
  }
}

// ---------------- 3x3 conv as implicit GEMM ----------------
// cat: bf16 (B, 512, H, W); Wc: [r][oc][ic] bf16; out: f32 (B, 256, H, W)
__global__ __launch_bounds__(128)
void conv3x3_kernel(const __bf16* __restrict__ cat, const __bf16* __restrict__ Wc,
                    float* __restrict__ out) {
  __shared__ __bf16 As[64 * 32];
  const int tid = threadIdx.x;
  const int lane = tid & 31;
  const int wid = tid >> 5;
  const int g = lane >> 4;
  const int ln = lane & 15;
  const int m0 = blockIdx.x * 64;
  const int n0 = blockIdx.y * 64;
  const int b = m0 >> 14;
  const int p0 = m0 & (HW_ - 1);

  v8f acc[4];
#pragma unroll
  for (int j = 0; j < 4; ++j) zero8(acc[j]);

  const int kseg = tid & 31;
  const int mseg = tid >> 5;

  for (int r = 0; r < 9; ++r) {
    const int dh = r / 3 - 1, dw = r % 3 - 1;
    for (int kc = 0; kc < 16; ++kc) {
      const __bf16* src = cat + ((size_t)(b * CCAT_ + kc * 32 + kseg)) * HW_;
#pragma unroll
      for (int i = 0; i < 16; ++i) {
        int mloc = mseg * 16 + i;
        int p = p0 + mloc;
        int h = (p >> 7) + dh;
        int w = (p & 127) + dw;
        __bf16 v = (__bf16)0.0f;
        if ((unsigned)h < 128u && (unsigned)w < 128u) v = src[h * 128 + w];
        As[mloc * 32 + kseg] = v;
      }
      __syncthreads();
      int mloc = wid * 16 + ln;
      v16bf a = load_arow(&As[mloc * 32], g);
#pragma unroll
      for (int nj = 0; nj < 4; ++nj) {
        int n = n0 + nj * 16 + ln;
        v16bf bb =
            load_frag16(Wc + ((size_t)r * E_ + n) * CCAT_ + kc * 32 + g * 16);
        acc[nj] = wmma_bf16(a, bb, acc[nj]);
      }
      __syncthreads();
    }
  }

#pragma unroll
  for (int nj = 0; nj < 4; ++nj)
#pragma unroll
    for (int i = 0; i < 8; ++i) {
      int mloc = wid * 16 + g * 8 + i;
      int p = p0 + mloc;
      int n = n0 + nj * 16 + ln;
      out[((size_t)(b * E_ + n)) * HW_ + p] = acc[nj][i];
    }
}

// ---------------- batchnorm (stats + apply/relu) ----------------
__global__ __launch_bounds__(256)
void bn_stats_kernel(const float* __restrict__ y, float* __restrict__ stats) {
  __shared__ float ssum[256], ssq[256];
  int c = blockIdx.x;
  float s = 0.0f, q = 0.0f;
  for (int j = threadIdx.x; j < PIX_; j += 256) {
    int b = j >> 14, p = j & (HW_ - 1);
    float v = y[((size_t)(b * E_ + c)) * HW_ + p];
    s += v;
    q += v * v;
  }
  ssum[threadIdx.x] = s;
  ssq[threadIdx.x] = q;
  __syncthreads();
  for (int st = 128; st; st >>= 1) {
    if (threadIdx.x < st) {
      ssum[threadIdx.x] += ssum[threadIdx.x + st];
      ssq[threadIdx.x] += ssq[threadIdx.x + st];
    }
    __syncthreads();
  }
  if (threadIdx.x == 0) {
    float mean = ssum[0] / (float)PIX_;
    float var = ssq[0] / (float)PIX_ - mean * mean;
    stats[c] = mean;
    stats[256 + c] = var;
  }
}

__global__ void bn_apply_kernel(float* __restrict__ y,
                                const float* __restrict__ stats,
                                const float* __restrict__ gamma,
                                const float* __restrict__ beta) {
  size_t i = (size_t)blockIdx.x * 256 + threadIdx.x;
  if (i >= (size_t)PIX_ * E_) return;
  int c = (int)((i >> 14) & 255);
  float m = stats[c], v = stats[256 + c];
  float val = (y[i] - m) * rsqrtf(v + BN_EPS_) * gamma[c] + beta[c];
  y[i] = fmaxf(val, 0.0f);
}

// ---------------- host orchestration ----------------
extern "C" void kernel_launch(void* const* d_in, const int* in_sizes, int n_in,
                              void* d_out, int out_size, void* d_ws, size_t ws_size,
                              hipStream_t stream) {
  (void)in_sizes; (void)n_in; (void)out_size; (void)ws_size;

  const float* x5  = (const float*)d_in[0];
  const float* d1  = (const float*)d_in[1];
  const float* d2  = (const float*)d_in[2];
  const float* kp  = (const float*)d_in[3];
  const float* qp  = (const float*)d_in[4];
  const float* vp  = (const float*)d_in[5];
  const float* sp  = (const float*)d_in[6];
  const float* ipw = (const float*)d_in[7];
  const float* ipb = (const float*)d_in[8];
  const float* opw = (const float*)d_in[9];
  const float* opb = (const float*)d_in[10];
  const float* pw  = (const float*)d_in[11];
  const float* gamma = (const float*)d_in[12];
  const float* beta  = (const float*)d_in[13];
  float* yout = (float*)d_out;

  char* wsb = (char*)d_ws;
  size_t off = 0;
  auto take = [&](size_t bytes) -> char* {
    char* p = wsb + off;
    off += (bytes + 255) & ~(size_t)255;
    return p;
  };
  float*  pe   = (float*)take((size_t)E_ * HW_ * 4);          // 16.8 MB
  __bf16* wq   = (__bf16*)take((size_t)256 * 256 * 2);
  __bf16* wk   = (__bf16*)take((size_t)256 * 512 * 2);
  __bf16* wv_  = (__bf16*)take((size_t)256 * 256 * 2);
  __bf16* wsp  = (__bf16*)take((size_t)256 * 256 * 2);
  __bf16* wip  = (__bf16*)take((size_t)768 * 256 * 2);
  __bf16* wop  = (__bf16*)take((size_t)256 * 256 * 2);
  __bf16* wcv  = (__bf16*)take((size_t)9 * 256 * 512 * 2);
  __bf16* q0   = (__bf16*)take((size_t)PIX_ * 256 * 2);       // 67 MB
  __bf16* k0   = (__bf16*)take((size_t)PIX_ * 256 * 2);
  __bf16* v0   = (__bf16*)take((size_t)PIX_ * 256 * 2);
  __bf16* qkv  = (__bf16*)take((size_t)PIX_ * 768 * 2);       // 201 MB
  __bf16* cat  = (__bf16*)take((size_t)B_ * CCAT_ * HW_ * 2); // 134 MB
  float*  stats = (float*)take(512 * 4);
  __bf16* obuf = q0;  // q0 is dead after in_proj; alias for attention output

  // ---- prep: bf16 weights, packed conv weight, positional encoding ----
  cvt_bf16_kernel<<<(65536 + 255) / 256, 256, 0, stream>>>(qp, wq, 65536);
  cvt_bf16_kernel<<<(131072 + 255) / 256, 256, 0, stream>>>(kp, wk, 131072);
  cvt_bf16_kernel<<<(65536 + 255) / 256, 256, 0, stream>>>(vp, wv_, 65536);
  cvt_bf16_kernel<<<(65536 + 255) / 256, 256, 0, stream>>>(sp, wsp, 65536);
  cvt_bf16_kernel<<<(196608 + 255) / 256, 256, 0, stream>>>(ipw, wip, 196608);
  cvt_bf16_kernel<<<(65536 + 255) / 256, 256, 0, stream>>>(opw, wop, 65536);
  pack_convw_kernel<<<(9 * 256 * 512 + 255) / 256, 256, 0, stream>>>(pw, wcv);
  pe_kernel<<<(E_ * HW_ + 255) / 256, 256, 0, stream>>>(pe);

  // ---- stage 1: q/k/v projections (+pe, windowed) and skip projection ----
  dim3 g1(PIX_ / 64, E_ / 64);
  gemm_colA_kernel<<<g1, 128, 0, stream>>>(d1, 256, wq, pe, q0, 0, 0);
  gemm_colA_kernel<<<g1, 128, 0, stream>>>(x5, 512, wk, pe, k0, 0, 0);
  gemm_colA_kernel<<<g1, 128, 0, stream>>>(d2, 256, wv_, pe, v0, 0, 0);
  gemm_colA_kernel<<<g1, 128, 0, stream>>>(d1, 256, wsp, nullptr, cat, 1, 256);

  // ---- stage 2: in_proj -> fused qkv buffer ----
  dim3 g2(PIX_ / 64, 256 / 64);
  gemm_rowA_kernel<<<g2, 128, 0, stream>>>(q0, 256, wip, ipb, qkv, 768, 0, 0);
  gemm_rowA_kernel<<<g2, 128, 0, stream>>>(k0, 256, wip + 65536, ipb + 256, qkv,
                                           768, 256, 0);
  gemm_rowA_kernel<<<g2, 128, 0, stream>>>(v0, 256, wip + 131072, ipb + 512, qkv,
                                           768, 512, 0);

  // ---- stage 3: windowed multi-head attention ----
  attn_kernel<<<NWIN_, 256, 0, stream>>>(qkv, obuf);

  // ---- stage 4: out_proj, window-reverse into concat buffer (channels 0..255) ----
  gemm_rowA_kernel<<<g2, 128, 0, stream>>>(obuf, 256, wop, opb, cat, 0, 0, 1);

  // ---- stage 5: 3x3 conv (implicit GEMM) ----
  dim3 g5(PIX_ / 64, E_ / 64);
  conv3x3_kernel<<<g5, 128, 0, stream>>>(cat, wcv, yout);

  // ---- stage 6: batchnorm + relu ----
  bn_stats_kernel<<<E_, 256, 0, stream>>>(yout, stats);
  bn_apply_kernel<<<(int)(((size_t)PIX_ * E_ + 255) / 256), 256, 0, stream>>>(
      yout, stats, gamma, beta);
}
